// myModel_2_50130858279885
// MI455X (gfx1250) — matrix operand order repaired and checked by reading, not compiled, not run on hardware
//
#include <hip/hip_runtime.h>
#include <math.h>

// ---------------------------------------------------------------------------
// MSR-Attention U-Net forward for MI455X (gfx1250), fp32 I/O, f16 WMMA convs.
// Conv layers run through an implicit-GEMM kernel:
//   TDM (tensor_load_to_lds, double-buffered) -> LDS weight tile ->
//   v_wmma_f32_16x16x32_f16, 4 waves/block sharing the weight tile.
// Depthwise Gaussian blurs / pool / upsample / SE / gating are VALU kernels.
// ---------------------------------------------------------------------------

typedef __attribute__((ext_vector_type(16))) _Float16 v16h;
typedef __attribute__((ext_vector_type(8)))  float    v8f;
typedef __attribute__((ext_vector_type(4)))  unsigned int tdm_v4u;
typedef __attribute__((ext_vector_type(8)))  int          tdm_v8i;
typedef __attribute__((ext_vector_type(4)))  int          tdm_v4i;

#if __has_include(<hip/amd_detail/amd_gfx1250_TDM.h>)
#define TDM_SIX_ARGS 1
#endif

static inline int ceil_div(int a, int b) { return (a + b - 1) / b; }

// ---------------------------------------------------------------------------
// TDM: async-load a 2-D f32 tile (32 elements x 16 rows) from global into LDS.
// tensor_dim0/1 give remaining K / remaining Cout so hardware OOB handling
// zero-fills the K-tail and Cout-tail of the tile for us.
// ---------------------------------------------------------------------------
static __device__ inline void tdm_load_w(unsigned int lds_off, const float* gsrc,
                                         int tdim0, int tdim1, int stride_elems)
{
  unsigned long long ga = (unsigned long long)(size_t)gsrc;
  tdm_v4u g0;
  g0.x = 1u;                                        // count=1 (valid), user mode
  g0.y = lds_off;                                   // LDS byte address
  g0.z = (unsigned int)ga;                          // global_addr[31:0]
  g0.w = (unsigned int)((ga >> 32) & 0x01FFFFFFu)   // global_addr[56:32]
       | (2u << 30);                                // type = 2 ("image")
  unsigned int d0 = (unsigned int)(tdim0 > 0 ? tdim0 : 0);
  unsigned int d1 = (unsigned int)(tdim1 > 0 ? tdim1 : 0);
  tdm_v8i g1;
  g1[0] = (int)(2u << 16);                          // workgroup_mask=0, data_size=4B
  g1[1] = (int)((d0 & 0xFFFFu) << 16);              // tensor_dim0[15:0] @ [31:16]
  g1[2] = (int)((d0 >> 16) | ((d1 & 0xFFFFu) << 16)); // dim0 hi | dim1 lo
  g1[3] = (int)((d1 >> 16) | (32u << 16));          // dim1 hi | tile_dim0 = 32
  g1[4] = (int)16u;                                 // tile_dim1 = 16, tile_dim2 = 0
  g1[5] = (int)(unsigned int)stride_elems;          // tensor_dim0_stride[31:0]
  g1[6] = 0;                                        // stride hi bits
  g1[7] = 0;
  tdm_v4i z4 = {0, 0, 0, 0};
#if defined(TDM_SIX_ARGS)
  tdm_v8i z8 = {0, 0, 0, 0, 0, 0, 0, 0};
  __builtin_amdgcn_tensor_load_to_lds(g0, g1, z4, z4, z8, 0);
#else
  __builtin_amdgcn_tensor_load_to_lds(g0, g1, z4, z4, 0);
#endif
}

// ---------------------------------------------------------------------------
// Implicit-GEMM conv (KS = 1 or 3, stride 1, pad KS/2) + bias + eval-BN + act.
// Block = 128 threads = 4 waves sharing one 16-cout tile; wave w owns output
// pixels [blockIdx.x*64 + w*16, +16).  K = Cin*KS*KS in chunks of 32; weight
// chunk staged to LDS by the TDM (double buffered, issued by wave 0, visible
// to all waves after s_wait_tensorcnt + barrier).  act: 0 none, 1 relu,
// 2 sigmoid.  inCtot/inCoff, outCtot/outCoff = channel slices for concat.
// ---------------------------------------------------------------------------
template <int KS>
__global__ __launch_bounds__(128)
void k_conv_wmma(const float* __restrict__ in, const float* __restrict__ wgt,
                 const float* __restrict__ cbias, const float* __restrict__ bng,
                 const float* __restrict__ bnb, float* __restrict__ out,
                 int Cin, int Cout, int H, int W, int act,
                 int inCtot, int inCoff, int outCtot, int outCoff)
{
  __shared__ float smw[2][16 * 32];     // double-buffered 16(cout) x 32(k) f32 tile

  const int tid   = threadIdx.x;
  const int lane  = tid & 31;
  const int wave  = tid >> 5;
  const int laneN = lane & 15;
  const int hi    = lane >> 4;          // half-wave selector per ISA fragment layout
  const int p0    = (blockIdx.x * 4 + wave) * 16;
  const int m0    = blockIdx.y * 16;
  const int n     = blockIdx.z;
  const int KK    = KS * KS;
  const int K     = Cin * KK;
  const int nK    = (K + 31) >> 5;
  const int HW    = H * W;

  const int  px     = p0 + laneN;       // this lane's output pixel (B/D column)
  const bool pvalid = px < HW;
  const int  oh     = pvalid ? px / W : 0;
  const int  ow     = pvalid ? px - oh * W : 0;

  const float* inb  = in + ((size_t)n * inCtot + inCoff) * (size_t)HW;
  const float* wrow = wgt + (size_t)m0 * K;
  const unsigned int lds0 = (unsigned int)(size_t)&smw[0][0];
  const unsigned int lds1 = (unsigned int)(size_t)&smw[1][0];

  if (wave == 0)                        // prologue: stage chunk 0
    tdm_load_w(lds0, wrow, K, Cout - m0, K);

  v8f acc = {};
  for (int kc = 0; kc < nK; ++kc) {
    if (wave == 0) __builtin_amdgcn_s_wait_tensorcnt(0);
    __syncthreads();                    // chunk kc visible to all waves

    const int kb = kc << 5;
    const int nb = kb + 32;
    if (wave == 0 && nb < K)            // overlap: stage chunk kc+1
      tdm_load_w((kc & 1) ? lds0 : lds1, wrow + nb, K - nb, Cout - m0, K);

    const float* wtile = &smw[kc & 1][laneN * 32];
    v16h af, bf;
#pragma unroll
    for (int i = 0; i < 16; ++i) {
      // A fragment (weights) from LDS, contiguous k order (same as B):
      //   lanes 0-15: K = 0..15; lanes 16-31: K = 16..31
      int ka = hi * 16 + i;
      af[i] = (_Float16)wtile[ka];      // TDM zero-filled past K / Cout edges

      // B fragment (im2col gather), contiguous k order (same as A):
      //   lanes 0-15: K = 0..15; lanes 16-31: K = 16..31
      int kbg = kb + hi * 16 + i;
      float xv = 0.f;
      if (pvalid && kbg < K) {
        int ci = kbg / KK;              // KS compile-time -> magic multiply
        int r  = kbg - ci * KK;
        int kh = r / KS;
        int kw = r - kh * KS;
        int ih = oh + kh - (KS >> 1);
        int iw = ow + kw - (KS >> 1);
        if ((unsigned)ih < (unsigned)H && (unsigned)iw < (unsigned)W)
          xv = inb[(size_t)ci * HW + ih * W + iw];
      }
      bf[i] = (_Float16)xv;
    }
    acc = __builtin_amdgcn_wmma_f32_16x16x32_f16(false, af, false, bf,
                                                 (short)0, acc, false, false);
  }

  // Epilogue: conv bias, eval-mode BN (running_mean=0, running_var=1), act.
  const float rs = rsqrtf(1.f + 1e-5f);
  float* outb = out + ((size_t)n * outCtot + outCoff) * (size_t)HW;
#pragma unroll
  for (int r = 0; r < 8; ++r) {
    int m = m0 + r + 8 * hi;            // D layout: M = vgpr + 8*(lane>=16)
    if (m < Cout && pvalid) {
      float v = acc[r];
      if (cbias) v += cbias[m];
      if (bng)   v = v * (bng[m] * rs) + bnb[m];
      if (act == 1)      v = fmaxf(v, 0.f);
      else if (act == 2) v = 1.f / (1.f + __expf(-v));
      outb[(size_t)m * HW + px] = v;
    }
  }
}

// ---------------------------------------------------------------------------
// Normalized 1-D Gaussian kernel table (k = 4s+1 taps).
// ---------------------------------------------------------------------------
__global__ void k_make_gauss(float* g, int ksz, float sigma)
{
  __shared__ float ssum[256];
  const int t = threadIdx.x;
  const float c = (ksz - 1) * 0.5f;
  float local = 0.f;
  for (int i = t; i < ksz; i += 256) {
    float d = (float)i - c;
    float v = __expf(-d * d / (2.f * sigma * sigma));
    g[i] = v;
    local += v;
  }
  ssum[t] = local;
  __syncthreads();
  for (int s = 128; s > 0; s >>= 1) {
    if (t < s) ssum[t] += ssum[t + s];
    __syncthreads();
  }
  const float inv = 1.f / ssum[0];
  for (int i = t; i < ksz; i += 256) g[i] *= inv;
}

// Depthwise vertical / horizontal Gaussian passes (zero pad == clamp range).
__global__ void k_gauss_v(const float* __restrict__ in, float* __restrict__ out,
                          const float* __restrict__ g, int R, int H, int W, int total)
{
  int i = blockIdx.x * blockDim.x + threadIdx.x;
  if (i >= total) return;
  int w = i % W;
  int h = (i / W) % H;
  size_t nc = (size_t)(i / (W * H));
  const float* b = in + nc * (size_t)(H * W);
  int t0 = -min(R, h), t1 = min(R, H - 1 - h);
  float s = 0.f;
  for (int t = t0; t <= t1; ++t) s += g[t + R] * b[(h + t) * W + w];
  out[i] = s;
}

__global__ void k_gauss_h(const float* __restrict__ in, float* __restrict__ out,
                          const float* __restrict__ g, int R, int H, int W, int total)
{
  int i = blockIdx.x * blockDim.x + threadIdx.x;
  if (i >= total) return;
  int w = i % W;
  size_t row = (size_t)(i / W);
  const float* b = in + row * (size_t)W;
  int t0 = -min(R, w), t1 = min(R, W - 1 - w);
  float s = 0.f;
  for (int t = t0; t <= t1; ++t) s += g[t + R] * b[w + t];
  out[i] = s;
}

// acc = (first ? 0 : acc) + (log1p(x) - log1p(sm)) / 3
__global__ void k_msr_term(float* __restrict__ acc, const float* __restrict__ x,
                           const float* __restrict__ sm, int total, int first)
{
  int i = blockIdx.x * blockDim.x + threadIdx.x;
  if (i >= total) return;
  float t = (log1pf(x[i]) - log1pf(sm[i])) * (1.f / 3.f);
  acc[i] = first ? t : acc[i] + t;
}

__global__ void k_maxpool2(const float* __restrict__ in, float* __restrict__ out,
                           int H, int W, int total)
{
  int i = blockIdx.x * blockDim.x + threadIdx.x;
  if (i >= total) return;
  int OW = W >> 1, OH = H >> 1;
  int ow = i % OW;
  int oh = (i / OW) % OH;
  size_t nc = (size_t)(i / (OW * OH));
  const float* b = in + nc * (size_t)(H * W) + (size_t)(2 * oh) * W + 2 * ow;
  out[i] = fmaxf(fmaxf(b[0], b[1]), fmaxf(b[W], b[W + 1]));
}

// Bilinear x2 upsample, align_corners=True; writes into channel slice of a
// larger (outCtot-channel) tensor at channel offset coff.
__global__ void k_upsample2(const float* __restrict__ in, float* __restrict__ out,
                            int C, int H, int W, int outCtot, int coff, int total)
{
  int i = blockIdx.x * blockDim.x + threadIdx.x;
  if (i >= total) return;
  int OH = 2 * H, OW = 2 * W;
  int ow = i % OW;
  int oh = (i / OW) % OH;
  int c  = (i / (OW * OH)) % C;
  int n  = i / (OW * OH * C);
  float ph = (OH > 1) ? oh * (float)(H - 1) / (float)(OH - 1) : 0.f;
  float pw = (OW > 1) ? ow * (float)(W - 1) / (float)(OW - 1) : 0.f;
  int hl = (int)ph; int hh = min(hl + 1, H - 1); float fh = ph - (float)hl;
  int wl = (int)pw; int wh = min(wl + 1, W - 1); float fw = pw - (float)wl;
  const float* b = in + ((size_t)n * C + c) * (size_t)(H * W);
  float v = b[hl * W + wl] * (1.f - fh) * (1.f - fw)
          + b[hl * W + wh] * (1.f - fh) * fw
          + b[hh * W + wl] * fh * (1.f - fw)
          + b[hh * W + wh] * fh * fw;
  out[((size_t)n * outCtot + coff + c) * (size_t)(OH * OW) + oh * OW + ow] = v;
}

// Global average pool: one block per (n,c).
__global__ void k_gap(const float* __restrict__ x, float* __restrict__ g, int HW)
{
  __shared__ float s[256];
  size_t base = (size_t)blockIdx.x * HW;
  float l = 0.f;
  for (int i = threadIdx.x; i < HW; i += 256) l += x[base + i];
  s[threadIdx.x] = l;
  __syncthreads();
  for (int st = 128; st > 0; st >>= 1) {
    if (threadIdx.x < st) s[threadIdx.x] += s[threadIdx.x + st];
    __syncthreads();
  }
  if (threadIdx.x == 0) g[blockIdx.x] = s[0] / (float)HW;
}

// SE FC stack: y1 = relu(W1 @ gap), sefac = sigmoid(W2 @ y1).
// Hd may be 0 (C=12) -> sefac = sigmoid(0) = 0.5, matching the reference.
__global__ void k_se_fc(const float* __restrict__ gap, const float* __restrict__ w1,
                        const float* __restrict__ w2, float* __restrict__ sefac,
                        int C, int Hd)
{
  __shared__ float y1[16];
  const int n = blockIdx.x, t = threadIdx.x;
  if (t < Hd) {
    float s = 0.f;
    for (int c = 0; c < C; ++c) s += w1[t * C + c] * gap[n * C + c];
    y1[t] = fmaxf(s, 0.f);
  }
  __syncthreads();
  for (int c = t; c < C; c += blockDim.x) {
    float s = 0.f;
    for (int h = 0; h < Hd; ++h) s += w2[c * Hd + h] * y1[h];
    sefac[n * C + c] = 1.f / (1.f + __expf(-s));
  }
}

// x = relu(h * sefac[n,c] + x)   (in-place on x)
__global__ void k_se_res(float* __restrict__ x, const float* __restrict__ h,
                         const float* __restrict__ sefac, int C, int HW, int total)
{
  int i = blockIdx.x * blockDim.x + threadIdx.x;
  if (i >= total) return;
  int c = (i / HW) % C;
  int n = i / (HW * C);
  x[i] = fmaxf(h[i] * sefac[n * C + c] + x[i], 0.f);
}

// a = relu(a + b)  (in-place on a)
__global__ void k_add_relu(float* __restrict__ a, const float* __restrict__ b, int total)
{
  int i = blockIdx.x * blockDim.x + threadIdx.x;
  if (i >= total) return;
  a[i] = fmaxf(a[i] + b[i], 0.f);
}

// out[n, coff+c, p] = x[n,c,p] * psi[n,0,p]  (psi broadcast over channels)
__global__ void k_mul_bcast(float* __restrict__ out, const float* __restrict__ x,
                            const float* __restrict__ psi, int C, int HW,
                            int outCtot, int coff, int total)
{
  int i = blockIdx.x * blockDim.x + threadIdx.x;
  if (i >= total) return;
  int p = i % HW;
  int c = (i / HW) % C;
  int n = i / (HW * C);
  out[((size_t)n * outCtot + coff + c) * (size_t)HW + p] = x[i] * psi[(size_t)n * HW + p];
}

// ---------------------------------------------------------------------------
// Host orchestration
// ---------------------------------------------------------------------------
struct CBRp { const float *w, *cb, *g, *b; };   // conv weight/bias, BN gamma/beta
struct SEp  { CBRp c1, c2; const float *w1, *w2; };
struct ATTp { CBRp wg, wx, psi; };

extern "C" void kernel_launch(void* const* d_in, const int* in_sizes, int n_in,
                              void* d_out, int out_size, void* d_ws, size_t ws_size,
                              hipStream_t stream)
{
  (void)out_size; (void)ws_size;
  const int B = 4;
  // Detect pytree flattening convention: JAX tree_flatten sorts dict keys
  // ('params' < 'x' -> x LAST); an insertion-order harness puts x FIRST.
  // x is the unique 2^18-element leaf (conv weights are co*ci*9, never 2^18).
  const bool xfirst = (n_in > 0 && in_sizes[0] == B * 256 * 256);

  int D[5], U[4], OUTW, OUTB, XI;
  if (xfirst) {
    XI = 0;
    int b = 1;
    for (int i = 0; i < 5; ++i) { D[i] = b; b += 24; }
    for (int i = 0; i < 4; ++i) { U[i] = b; b += 20; }
    OUTW = b; OUTB = b + 1;                    // insertion: conv {w, b}
  } else {
    for (int i = 0; i < 5; ++i) D[i] = i * 24; // sorted: d1..d5, out, u1..u4, x
    OUTB = 120; OUTW = 121;                    // sorted: conv {b, w}
    for (int i = 0; i < 4; ++i) U[i] = 122 + i * 20;
    XI = 202;
  }

  auto F = [&](int i) -> const float* { return (const float*)d_in[i]; };

  auto cbr_at = [&](int base) -> CBRp {
    CBRp r;
    if (xfirst) { r.w = F(base); r.cb = F(base + 1); r.g = F(base + 2); r.b = F(base + 3); }
    else        { r.b = F(base); r.g = F(base + 1); r.cb = F(base + 2); r.w = F(base + 3); }
    return r;
  };
  auto seres_at = [&](int s) -> SEp {
    SEp r;
    if (xfirst) {
      r.c1.w = F(s + 0); r.c1.cb = F(s + 1); r.c1.g = F(s + 2); r.c1.b = F(s + 3);
      r.c2.w = F(s + 4); r.c2.cb = F(s + 5); r.c2.g = F(s + 6); r.c2.b = F(s + 7);
    } else {
      r.c1.b = F(s + 0); r.c1.g = F(s + 1); r.c2.b = F(s + 2); r.c2.g = F(s + 3);
      r.c1.cb = F(s + 4); r.c1.w = F(s + 5); r.c2.cb = F(s + 6); r.c2.w = F(s + 7);
    }
    r.w1 = F(s + 8); r.w2 = F(s + 9);
    return r;
  };
  auto att_at = [&](int u) -> ATTp {
    ATTp r;
    if (xfirst) {
      r.wg.w = F(u + 0); r.wg.cb = F(u + 1); r.wg.g = F(u + 2); r.wg.b = F(u + 3);
      r.wx.w = F(u + 4); r.wx.cb = F(u + 5); r.wx.g = F(u + 6); r.wx.b = F(u + 7);
      r.psi.w = F(u + 8); r.psi.cb = F(u + 9); r.psi.g = F(u + 10); r.psi.b = F(u + 11);
    } else {
      r.wg.b = F(u + 0); r.wg.g = F(u + 1);      // bng
      r.psi.b = F(u + 2); r.psi.g = F(u + 3);    // bnp
      r.wx.b = F(u + 4); r.wx.g = F(u + 5);      // bnx
      r.psi.cb = F(u + 6); r.psi.w = F(u + 7);   // psi conv
      r.wg.cb = F(u + 8); r.wg.w = F(u + 9);     // wg conv
      r.wx.cb = F(u + 10); r.wx.w = F(u + 11);   // wx conv
    }
    return r;
  };

  // ---- workspace bump allocator (~125 MB fp32 total) ----------------------
  float* ws = (float*)d_ws;
  size_t off = 0;
  auto alloc = [&](size_t n) -> float* {
    float* p = ws + off;
    off += (n + 63) & ~(size_t)63;
    return p;
  };
  float* gk0 = alloc(61);
  float* gk1 = alloc(321);
  float* gk2 = alloc(1001);
  float* x1 = alloc((size_t)B * 12 * 256 * 256);
  float* x2 = alloc((size_t)B * 24 * 128 * 128);
  float* x3 = alloc((size_t)B * 48 * 64 * 64);
  float* x4 = alloc((size_t)B * 96 * 32 * 32);
  float* x5 = alloc((size_t)B * 192 * 16 * 16);
  float* y1 = alloc((size_t)B * 96 * 32 * 32);
  float* y2 = alloc((size_t)B * 48 * 64 * 64);
  float* y3 = alloc((size_t)B * 24 * 128 * 128);
  float* y4 = alloc((size_t)B * 12 * 256 * 256);
  float* bt1   = alloc((size_t)B * 12 * 256 * 256);   // generic temp A
  float* bt2   = alloc((size_t)B * 12 * 256 * 256);   // generic temp B
  float* m_acc = alloc((size_t)B * 12 * 256 * 256);   // MSR accumulator
  float* psi   = alloc((size_t)B * 256 * 256);
  float* gap   = alloc((size_t)B * 192);
  float* sefac = alloc((size_t)B * 192);
  float* cc    = alloc((size_t)B * 36 * 256 * 256);   // concat buffer (max 36ch@256^2)

  auto ewg = [&](int n) { return dim3((unsigned)ceil_div(n, 256)); };

  auto conv = [&](const float* in, const CBRp& p, float* out, int Cin, int Cout,
                  int H, int W, int ks, int act, int inCtot, int inCoff,
                  int outCtot, int outCoff) {
    dim3 grid((unsigned)ceil_div(H * W, 64), (unsigned)ceil_div(Cout, 16), (unsigned)B);
    if (ks == 1)
      k_conv_wmma<1><<<grid, dim3(128), 0, stream>>>(in, p.w, p.cb, p.g, p.b, out,
          Cin, Cout, H, W, act, inCtot, inCoff, outCtot, outCoff);
    else
      k_conv_wmma<3><<<grid, dim3(128), 0, stream>>>(in, p.w, p.cb, p.g, p.b, out,
          Cin, Cout, H, W, act, inCtot, inCoff, outCtot, outCoff);
  };

  auto seres_run = [&](float* x, int sbase, int C, int H, int W) {
    SEp p = seres_at(sbase);
    conv(x,   p.c1, bt1, C, C, H, W, 3, 1, C, 0, C, 0);
    conv(bt1, p.c2, bt2, C, C, H, W, 3, 0, C, 0, C, 0);
    int HW = H * W, n = B * C * HW;
    k_gap<<<dim3((unsigned)(B * C)), dim3(256), 0, stream>>>(bt2, gap, HW);
    k_se_fc<<<dim3((unsigned)B), dim3(256), 0, stream>>>(gap, p.w1, p.w2, sefac, C, C / 16);
    k_se_res<<<ewg(n), dim3(256), 0, stream>>>(x, bt2, sefac, C, HW, n);
  };

  auto down_run = [&](const float* in, float* out, int base, int Cin, int Cout,
                      int Hin, int Win, bool pool) {
    const float* src = in;
    int H = Hin, W = Win;
    if (pool) {
      H >>= 1; W >>= 1;
      int n = B * Cin * H * W;
      k_maxpool2<<<ewg(n), dim3(256), 0, stream>>>(in, bt1, Hin, Win, n);
      src = bt1;
    }
    conv(src, cbr_at(base), out, Cin, Cout, H, W, 3, 1, Cin, 0, Cout, 0);
    seres_run(out, base + 4,  Cout, H, W);
    seres_run(out, base + 14, Cout, H, W);
  };

  auto msr_run = [&](const float* x, float* acc, int C, int H, int W) {
    const int    SC[3] = {15, 80, 250};
    const float* GK[3] = {gk0, gk1, gk2};
    int n = B * C * H * W;
    for (int i = 0; i < 3; ++i) {
      int R = 2 * SC[i];
      k_gauss_v<<<ewg(n), dim3(256), 0, stream>>>(x,   bt1, GK[i], R, H, W, n);
      k_gauss_h<<<ewg(n), dim3(256), 0, stream>>>(bt1, bt2, GK[i], R, H, W, n);
      k_msr_term<<<ewg(n), dim3(256), 0, stream>>>(acc, x, bt2, n, i == 0 ? 1 : 0);
    }
  };

  auto up_run = [&](const float* xs, const float* xd, float* y, int base,
                    int C1, int C2, int H1, int W1) {
    int Ctot = C1 + C2;
    // upsample deep feature into concat channels [C1, Ctot)
    int nu = B * C2 * H1 * W1;
    k_upsample2<<<ewg(nu), dim3(256), 0, stream>>>(xd, cc, C2, H1 >> 1, W1 >> 1, Ctot, C1, nu);
    // attention gate
    ATTp ap = att_at(base);
    msr_run(xs, m_acc, C1, H1, W1);
    conv(cc,    ap.wg, bt1, C2, C1, H1, W1, 1, 0, Ctot, C1, C1, 0);  // a = BN(W_g g)
    conv(m_acc, ap.wx, bt2, C1, C1, H1, W1, 1, 0, C1, 0, C1, 0);     // b = BN(W_x m)
    int na = B * C1 * H1 * W1;
    k_add_relu<<<ewg(na), dim3(256), 0, stream>>>(bt1, bt2, na);     // relu(a+b)
    conv(bt1, ap.psi, psi, C1, 1, H1, W1, 1, 2, C1, 0, 1, 0);        // sigmoid(BN(psi))
    k_mul_bcast<<<ewg(na), dim3(256), 0, stream>>>(cc, xs, psi, C1, H1 * W1, Ctot, 0, na);
    // concat -> conv -> conv
    conv(cc,  cbr_at(base + 12), bt1, Ctot, C1, H1, W1, 3, 1, Ctot, 0, C1, 0);
    conv(bt1, cbr_at(base + 16), y,   C1,   C1, H1, W1, 3, 1, C1, 0, C1, 0);
  };

  // ---- forward pass -------------------------------------------------------
  k_make_gauss<<<dim3(1), dim3(256), 0, stream>>>(gk0, 61,   15.f);
  k_make_gauss<<<dim3(1), dim3(256), 0, stream>>>(gk1, 321,  80.f);
  k_make_gauss<<<dim3(1), dim3(256), 0, stream>>>(gk2, 1001, 250.f);

  const float* xin = F(XI);
  down_run(xin, x1, D[0], 1,   12,  256, 256, false);
  down_run(x1,  x2, D[1], 12,  24,  256, 256, true);
  down_run(x2,  x3, D[2], 24,  48,  128, 128, true);
  down_run(x3,  x4, D[3], 48,  96,  64,  64,  true);
  down_run(x4,  x5, D[4], 96,  192, 32,  32,  true);

  up_run(x4, x5, y1, U[0], 96, 192, 32,  32);
  up_run(x3, y1, y2, U[1], 48, 96,  64,  64);
  up_run(x2, y2, y3, U[2], 24, 48,  128, 128);
  up_run(x1, y3, y4, U[3], 12, 24,  256, 256);

  CBRp po; po.w = F(OUTW); po.cb = F(OUTB); po.g = nullptr; po.b = nullptr;
  conv(y4, po, (float*)d_out, 12, 1, 256, 256, 3, 0, 12, 0, 1, 0);
}
